// STGNNMixer_49667001811055
// MI455X (gfx1250) — matrix operand order repaired and checked by reading, not compiled, not run on hardware
//
#include <hip/hip_runtime.h>
#include <hip/hip_bf16.h>
#include <math.h>

typedef __attribute__((ext_vector_type(16))) __bf16 v16bf;
typedef __attribute__((ext_vector_type(8)))  float  v8f;

#define B_   8
#define N_   256
#define S_   96
#define FIN_ 8
#define H_   128
#define P_   24
#define FUT_ 8
#define NBLK_ 3

// ---------------- WMMA helpers (CDNA5 16x16x32 bf16, fp32 accum) ----------------

__device__ __forceinline__ v8f wmma_bf(v16bf a, v16bf b, v8f c) {
  return __builtin_amdgcn_wmma_f32_16x16x32_bf16(false, a, false, b, (short)0, c,
                                                 false, false);
}

__device__ __forceinline__ v8f zero8() {
  v8f c;
#pragma unroll
  for (int i = 0; i < 8; ++i) c[i] = 0.f;
  return c;
}

// A-matrix 16x32 bf16 fragment from row-major bf16 buffer (tile origin given).
// Contiguous k = 16h + s order (same as the B loaders): lanes 0-15 -> M=lane, K 0..15; lanes 16-31 -> K 16..31.
// Both 8-element runs are contiguous 16B -> ds_load_b128 / global b128.
__device__ __forceinline__ v16bf loadA_bf(const __bf16* A, int lda, int lane) {
  int m  = lane & 15;
  int kb = (lane < 16) ? 0 : 16;
  const __bf16* p = A + (size_t)m * lda + kb;
  v16bf a;
#pragma unroll
  for (int i = 0; i < 8; ++i) { a[i] = p[i]; a[i + 8] = p[8 + i]; }
  return a;
}

// B-matrix 32x16 fragment from TRANSPOSED buffer BT[n][k] (K contiguous, stride ldk).
// lanes 0-15 -> N=lane, K 0..15; lanes 16-31 -> K 16..31.
// 16 contiguous bf16 = 32B aligned -> two b128 loads, no packing movs.
__device__ __forceinline__ v16bf loadBT_bf(const __bf16* BT, int ldk, int lane) {
  int n  = lane & 15;
  int kb = (lane < 16) ? 0 : 16;
  const __bf16* p = BT + (size_t)n * ldk + kb;
  v16bf b;
#pragma unroll
  for (int i = 0; i < 16; ++i) b[i] = p[i];
  return b;
}

// Same as loadBT_bf but from fp32 source with on-the-fly conversion:
// logical B[k][n] = M[col0+n][k0+k] for row-major M (M^T access, contiguous reads).
__device__ __forceinline__ v16bf loadBt_f32(const float* M, int ldm, int col0, int k0,
                                            int lane) {
  int n  = lane & 15;
  int kb = k0 + ((lane < 16) ? 0 : 16);
  const float* p = M + (size_t)(col0 + n) * ldm + kb;
  v16bf b;
#pragma unroll
  for (int i = 0; i < 16; ++i) b[i] = (__bf16)p[i];
  return b;
}

// ---------------- K1: instance norm + projection + node emb + ctx pool ----------------
// grid = B*N blocks, 128 threads (thread = output channel h)
__global__ void k_norm_proj_ctx(const float* __restrict__ x, const float* __restrict__ in_w,
                                const float* __restrict__ in_b, const float* __restrict__ projW,
                                const float* __restrict__ projb, const float* __restrict__ node_emb,
                                const float* __restrict__ ctxW, const float* __restrict__ ctxb,
                                float* __restrict__ h, float* __restrict__ ctx) {
  int bn = blockIdx.x;
  int n  = bn & (N_ - 1);
  int tid = threadIdx.x;
  __shared__ float xs[S_ * FIN_];
  __shared__ float mu[FIN_], rstd[FIN_];
  const float* xp = x + (size_t)bn * S_ * FIN_;
  for (int i = tid; i < S_ * FIN_; i += 128) xs[i] = xp[i];
  __syncthreads();
  if (tid < FIN_) {
    float m = 0.f;
    for (int s = 0; s < S_; ++s) m += xs[s * FIN_ + tid];
    m *= (1.0f / S_);
    float v = 0.f;
    for (int s = 0; s < S_; ++s) { float d = xs[s * FIN_ + tid] - m; v += d * d; }
    v *= (1.0f / S_);
    mu[tid] = m;
    rstd[tid] = rsqrtf(v + 1e-5f);
  }
  __syncthreads();
  int hh = tid;
  float pw[FIN_], sc[FIN_], of[FIN_];
#pragma unroll
  for (int f = 0; f < FIN_; ++f) {
    pw[f] = projW[f * H_ + hh];
    sc[f] = rstd[f] * in_w[f];
    of[f] = in_b[f] - mu[f] * sc[f];
  }
  float base = projb[hh] + node_emb[n * H_ + hh];
  float acc = 0.f;
  float* hp = h + (size_t)bn * S_ * H_;
  for (int s = 0; s < S_; ++s) {
    float v = base;
#pragma unroll
    for (int f = 0; f < FIN_; ++f) v += (xs[s * FIN_ + f] * sc[f] + of[f]) * pw[f];
    hp[s * H_ + hh] = v;
    acc += v * ctxW[s];
  }
  ctx[bn * H_ + hh] = acc + ctxb[0];
}

// ---------------- K2: Y = X @ W + b, X[2048,128], W[128,128] (WMMA) ----------------
// grid = 128 row-tiles, 256 threads (8 waves, one 16-col tile each)
__global__ void k_linear128(const float* __restrict__ X, const float* __restrict__ W,
                            const float* __restrict__ bvec, float* __restrict__ Y) {
  __shared__ __bf16 xs[16 * H_];     // A strip [m][k]
  __shared__ __bf16 wsT[H_ * H_];    // B transposed: wsT[n][k] = W[k][n]
  int tid = threadIdx.x, wave = tid >> 5, lane = tid & 31;
  int row0 = blockIdx.x * 16;
  for (int i = tid; i < 16 * H_; i += 256) xs[i] = (__bf16)X[(size_t)row0 * H_ + i];
  for (int i = tid; i < H_ * H_; i += 256) {
    int nn = i >> 7, kk = i & (H_ - 1);
    wsT[i] = (__bf16)W[kk * H_ + nn];
  }
  __syncthreads();
  int col0 = wave * 16;
  v8f c = zero8();
#pragma unroll
  for (int k0 = 0; k0 < H_; k0 += 32) {
    v16bf a = loadA_bf(xs + k0, H_, lane);
    v16bf b = loadBT_bf(wsT + (size_t)col0 * H_ + k0, H_, lane);
    c = wmma_bf(a, b, c);
  }
  int col = col0 + (lane & 15);
  int r0 = (lane < 16) ? 0 : 8;
  float bb = bvec[col];
#pragma unroll
  for (int r = 0; r < 8; ++r) Y[(size_t)(row0 + r0 + r) * H_ + col] = c[r] + bb;
}

// ---------------- K3a: dyn = relu(Q K^T / sqrt(H)) (WMMA) ----------------
// grid = B*16 blocks (b, 16-row tile), 256 threads
__global__ void k_dyn(const float* __restrict__ Qm, const float* __restrict__ Km,
                      float* __restrict__ dyn) {
  int b = blockIdx.x >> 4, nt = blockIdx.x & 15;
  __shared__ __bf16 qs[16 * H_];
  int tid = threadIdx.x, wave = tid >> 5, lane = tid & 31;
  const float* Qb = Qm + ((size_t)b * N_ + nt * 16) * H_;
  for (int i = tid; i < 16 * H_; i += 256) qs[i] = (__bf16)Qb[i];
  __syncthreads();
  const float* Kb = Km + (size_t)b * N_ * H_;
  const float scale = 0.08838834764831845f;  // 1/sqrt(128)
  for (int ct = wave; ct < 16; ct += 8) {
    int col0 = ct * 16;
    v8f c = zero8();
#pragma unroll
    for (int k0 = 0; k0 < H_; k0 += 32) {
      v16bf a = loadA_bf(qs + k0, H_, lane);
      v16bf bB = loadBt_f32(Kb, H_, col0, k0, lane);   // K rows are contiguous
      c = wmma_bf(a, bB, c);
    }
    int col = col0 + (lane & 15);
    int r0 = (lane < 16) ? 0 : 8;
#pragma unroll
    for (int r = 0; r < 8; ++r) {
      float v = c[r] * scale;
      dyn[((size_t)b * N_ + nt * 16 + r0 + r) * N_ + col] = v > 0.f ? v : 0.f;
    }
  }
}

__device__ __forceinline__ float blocksum256(float v, float* red, int tid) {
  red[tid] = v;
  __syncthreads();
  for (int s = 128; s > 0; s >>= 1) {
    if (tid < s) red[tid] += red[tid + s];
    __syncthreads();
  }
  float r = red[0];
  __syncthreads();
  return r;
}

// ---------------- K3b: top-5 mask + softmax + blend + L1 norm + diag + dis ----------------
// grid = B*N blocks (one row), 256 threads (thread = column m)
__global__ void k_rowgraph(float* __restrict__ dynA, const float* __restrict__ adj,
                           const float* __restrict__ galpha, float* __restrict__ dis) {
  int row = blockIdx.x;          // b*N + n
  int n = row & (N_ - 1);
  int tid = threadIdx.x;         // m
  __shared__ float sv[256];
  __shared__ int si[256];
  float* rp = dynA + (size_t)row * N_;
  float v = rp[tid];             // relu'd score
  float cur = v;
  bool picked = false;
  float m1 = 0.f;
#pragma unroll
  for (int it = 0; it < 5; ++it) {
    sv[tid] = cur; si[tid] = tid;
    __syncthreads();
    for (int s = 128; s > 0; s >>= 1) {
      if (tid < s) {
        if (sv[tid + s] > sv[tid] ||
            (sv[tid + s] == sv[tid] && si[tid + s] < si[tid])) {
          sv[tid] = sv[tid + s]; si[tid] = si[tid + s];
        }
      }
      __syncthreads();
    }
    int widx = si[0];
    float wval = sv[0];
    __syncthreads();
    if (it == 0) m1 = wval;
    if (tid == widx) { picked = true; cur = -1e30f; }
    __syncthreads();
  }
  float e = picked ? expf(v - m1) : 0.f;
  float esum = blocksum256(e, sv, tid);
  float soft = e / esum;
  float ga = galpha[0];
  float alpha = 1.f / (1.f + expf(-ga));
  float comb = alpha * adj[(size_t)n * N_ + tid] + (1.f - alpha) * soft;
  float l1 = blocksum256(fabsf(comb), sv, tid);
  comb /= fmaxf(l1, 1e-12f);
  if (tid == n) comb = 1.0f;                 // diag set AFTER L1 norm
  float rowsum = blocksum256(comb, sv, tid);
  rp[tid] = comb;
  if (tid == 0) dis[row] = rsqrtf(fmaxf(rowsum, 1.0f));
}

// ---------------- K3c: An = dis[n] * A * dis[m] ----------------
__global__ void k_an(const float* __restrict__ A, const float* __restrict__ dis,
                     float* __restrict__ An) {
  int row = blockIdx.x;          // b*N + n
  int b = row >> 8;
  int tid = threadIdx.x;
  float dn = dis[row];
  float dm = dis[b * N_ + tid];
  An[(size_t)row * N_ + tid] = dn * A[(size_t)row * N_ + tid] * dm;
}

// ---------------- K4: temporal mix + exact GELU + residual (WMMA) ----------------
// grid = B*N blocks (one (b,n) tile), 256 threads
__global__ void k_temporal(float* __restrict__ h, const float* __restrict__ Wt,
                           const float* __restrict__ bt) {
  __shared__ __bf16 hbT[H_ * S_];  // B transposed: hbT[hh][s]  24KB
  __shared__ __bf16 wb[S_ * S_];   // A matrix [t][s]           18KB
  int bn = blockIdx.x, tid = threadIdx.x, wave = tid >> 5, lane = tid & 31;
  float* hp = h + (size_t)bn * S_ * H_;
  for (int i = tid; i < S_ * H_; i += 256) {
    int s = i >> 7, hh = i & (H_ - 1);
    hbT[hh * S_ + s] = (__bf16)hp[i];
  }
  for (int i = tid; i < S_ * S_; i += 256) wb[i] = (__bf16)Wt[i];
  __syncthreads();
  for (int j = 0; j < 6; ++j) {            // 6 row tiles of t
    v8f c = zero8();
#pragma unroll
    for (int k0 = 0; k0 < S_; k0 += 32) {  // K = S
      v16bf a = loadA_bf(wb + (size_t)(j * 16) * S_ + k0, S_, lane);
      v16bf b = loadBT_bf(hbT + (size_t)(wave * 16) * S_ + k0, S_, lane);
      c = wmma_bf(a, b, c);
    }
    int col = wave * 16 + (lane & 15);
    int r0 = (lane < 16) ? 0 : 8;
#pragma unroll
    for (int r = 0; r < 8; ++r) {
      int trow = j * 16 + r0 + r;
      float t = c[r] + bt[trow];
      float g = 0.5f * t * (1.0f + erff(t * 0.70710678118654752f));  // exact GELU
      size_t off = (size_t)trow * H_ + col;
      hp[off] = hp[off] + g;               // residual (hbT snapshot unaffected)
    }
  }
}

// ---------------- K5: hwT[col][m] = (h @ gcn_W) transposed -> bf16 scratch (WMMA) ----------------
// grid = B*N blocks, 256 threads. Output stored node-minor for the spatial kernel.
__global__ void k_gcnlin(const float* __restrict__ h, const float* __restrict__ Wg,
                         __bf16* __restrict__ hwT) {
  __shared__ __bf16 hb[S_ * H_];   // A [s][h] 24KB
  __shared__ __bf16 wgT[H_ * H_];  // B transposed: wgT[g][h] 32KB
  int bn = blockIdx.x, tid = threadIdx.x, wave = tid >> 5, lane = tid & 31;
  int b = bn >> 8, m = bn & (N_ - 1);
  const float* hp = h + (size_t)bn * S_ * H_;
  for (int i = tid; i < S_ * H_; i += 256) hb[i] = (__bf16)hp[i];
  for (int i = tid; i < H_ * H_; i += 256) {
    int g = i >> 7, hk = i & (H_ - 1);
    wgT[i] = (__bf16)Wg[hk * H_ + g];
  }
  __syncthreads();
  __bf16* op = hwT + (size_t)b * S_ * H_ * N_ + m;
  for (int j = 0; j < 6; ++j) {
    v8f c = zero8();
#pragma unroll
    for (int k0 = 0; k0 < H_; k0 += 32) {
      v16bf a = loadA_bf(hb + (size_t)(j * 16) * H_ + k0, H_, lane);
      v16bf b = loadBT_bf(wgT + (size_t)(wave * 16) * H_ + k0, H_, lane);
      c = wmma_bf(a, b, c);
    }
    int col = wave * 16 + (lane & 15);
    int r0 = (lane < 16) ? 0 : 8;
#pragma unroll
    for (int r = 0; r < 8; ++r) {
      int srow = j * 16 + r0 + r;
      op[(size_t)(srow * H_ + col) * N_] = (__bf16)c[r];
    }
  }
}

// ---------------- K6: h += An @ hw + bias (WMMA over nodes) ----------------
// grid = B*16 blocks (b, 16-row node tile), 256 threads; cols = S*H = 12288
// hwT layout [b][col][m] makes every B fragment two contiguous b128 loads.
__global__ void k_spatial(float* __restrict__ h, const __bf16* __restrict__ hwT,
                          const float* __restrict__ An, const float* __restrict__ gb) {
  int b = blockIdx.x >> 4, nt = blockIdx.x & 15;
  __shared__ __bf16 as[16 * N_];   // An strip bf16, 8KB
  int tid = threadIdx.x, wave = tid >> 5, lane = tid & 31;
  const float* Ab = An + ((size_t)b * N_ + nt * 16) * N_;
  for (int i = tid; i < 16 * N_; i += 256) as[i] = (__bf16)Ab[i];
  __syncthreads();
  const int COLS = S_ * H_;                // 12288
  const __bf16* hwb = hwT + (size_t)b * COLS * N_;
  float* hp = h + ((size_t)b * N_ + nt * 16) * COLS;
  for (int ct = wave; ct < COLS / 16; ct += 8) {
    int col0 = ct * 16;
    if (ct + 8 < COLS / 16)                // prefetch next strip (global_prefetch_b8)
      __builtin_prefetch(hwb + (size_t)(col0 + 128) * N_ + (lane & 15) * N_, 0, 0);
    v8f c = zero8();
#pragma unroll
    for (int k0 = 0; k0 < N_; k0 += 32) {  // K = nodes m
      v16bf a = loadA_bf(as + k0, N_, lane);
      v16bf bB = loadBT_bf(hwb + (size_t)col0 * N_ + k0, N_, lane);
      c = wmma_bf(a, bB, c);
    }
    int col = col0 + (lane & 15);
    float bias = gb[col & (H_ - 1)];
    int r0 = (lane < 16) ? 0 : 8;
#pragma unroll
    for (int r = 0; r < 8; ++r) {
      size_t off = (size_t)(r0 + r) * COLS + col;
      hp[off] = hp[off] + c[r] + bias;     // z = h + sp + b
    }
  }
}

// ---------------- K7: LayerNorm over H per (b,n,s) ----------------
// grid = B*N*S blocks, 128 threads
__global__ void k_ln(float* __restrict__ h, const float* __restrict__ g,
                     const float* __restrict__ bta) {
  size_t row = blockIdx.x;
  int t = threadIdx.x;
  __shared__ float red[128];
  float v = h[row * H_ + t];
  red[t] = v;
  __syncthreads();
  for (int s = 64; s > 0; s >>= 1) { if (t < s) red[t] += red[t + s]; __syncthreads(); }
  float m = red[0] * (1.0f / H_);
  __syncthreads();
  float d = v - m;
  red[t] = d * d;
  __syncthreads();
  for (int s = 64; s > 0; s >>= 1) { if (t < s) red[t] += red[t + s]; __syncthreads(); }
  float var = red[0] * (1.0f / H_);
  __syncthreads();
  h[row * H_ + t] = d * rsqrtf(var + 1e-5f) * g[t] + bta[t];
}

// ---------------- K8: head (hist ++ future proj) @ head_W, relu ----------------
// grid = B*N blocks, 256 threads
__global__ void k_head(const float* __restrict__ h, const float* __restrict__ xf,
                       const float* __restrict__ fW, const float* __restrict__ fb,
                       const float* __restrict__ hW, const float* __restrict__ hb,
                       float* __restrict__ out) {
  int bn = blockIdx.x, tid = threadIdx.x;
  float acc[P_];
#pragma unroll
  for (int p = 0; p < P_; ++p) acc[p] = 0.f;
  const float* hp = h + (size_t)bn * S_ * H_;
  for (int idx = tid; idx < S_ * H_; idx += 256) {
    float v = hp[idx];
    const float* w = hW + (size_t)idx * P_;
#pragma unroll
    for (int p = 0; p < P_; ++p) acc[p] += v * w[p];
  }
  const float* xfp = xf + (size_t)bn * P_ * FUT_;
  for (int idx = tid; idx < P_ * H_; idx += 256) {
    int pp = idx >> 7, hh = idx & (H_ - 1);
    float v = fb[hh];
#pragma unroll
    for (int f = 0; f < FUT_; ++f) v += xfp[pp * FUT_ + f] * fW[f * H_ + hh];
    const float* w = hW + (size_t)(S_ * H_ + idx) * P_;
#pragma unroll
    for (int p = 0; p < P_; ++p) acc[p] += v * w[p];
  }
  __shared__ float red[256];
  for (int p = 0; p < P_; ++p) {
    red[tid] = acc[p];
    __syncthreads();
    for (int s = 128; s > 0; s >>= 1) { if (tid < s) red[tid] += red[tid + s]; __syncthreads(); }
    if (tid == 0) {
      float o = red[0] + hb[p];
      out[(size_t)bn * P_ + p] = o > 0.f ? o : 0.f;
    }
    __syncthreads();
  }
}

// ---------------- launch ----------------
extern "C" void kernel_launch(void* const* d_in, const int* in_sizes, int n_in,
                              void* d_out, int out_size, void* d_ws, size_t ws_size,
                              hipStream_t stream) {
  const float* x        = (const float*)d_in[0];
  const float* x_future = (const float*)d_in[1];
  const float* adj      = (const float*)d_in[2];
  const float* in_w     = (const float*)d_in[3];
  const float* in_b     = (const float*)d_in[4];
  const float* proj_W   = (const float*)d_in[5];
  const float* proj_b   = (const float*)d_in[6];
  const float* node_emb = (const float*)d_in[7];
  const float* ctx_W    = (const float*)d_in[8];
  const float* ctx_b    = (const float*)d_in[9];
  const float* q_W      = (const float*)d_in[10];
  const float* q_b      = (const float*)d_in[11];
  const float* k_W      = (const float*)d_in[12];
  const float* k_b      = (const float*)d_in[13];
  const float* g_alpha  = (const float*)d_in[14];
  const float* tW       = (const float*)d_in[15];
  const float* tb       = (const float*)d_in[16];
  const float* gW       = (const float*)d_in[17];
  const float* gb       = (const float*)d_in[18];
  const float* lng      = (const float*)d_in[19];
  const float* lnb      = (const float*)d_in[20];
  const float* fW       = (const float*)d_in[21];
  const float* fb       = (const float*)d_in[22];
  const float* hW       = (const float*)d_in[23];
  const float* hb       = (const float*)d_in[24];
  float* out = (float*)d_out;

  char* w = (char*)d_ws;
  const size_t BN   = (size_t)B_ * N_;
  const size_t HSZ  = BN * S_ * H_;              // 25,165,824 elems
  float*  h    = (float*)w;            w += HSZ * 4;         // 96 MB
  __bf16* hwT  = (__bf16*)w;           w += HSZ * 2;         // 48 MB, [b][s*H+g][m]
  float*  ctx  = (float*)w;            w += BN * H_ * 4;
  float*  Qm   = (float*)w;            w += BN * H_ * 4;
  float*  Km   = (float*)w;            w += BN * H_ * 4;
  float*  dynA = (float*)w;            w += BN * N_ * 4;     // dyn, then A
  float*  An   = (float*)w;            w += BN * N_ * 4;
  float*  dis  = (float*)w;            w += BN * 4;

  k_norm_proj_ctx<<<(int)BN, 128, 0, stream>>>(x, in_w, in_b, proj_W, proj_b,
                                               node_emb, ctx_W, ctx_b, h, ctx);
  k_linear128<<<(int)(BN / 16), 256, 0, stream>>>(ctx, q_W, q_b, Qm);
  k_linear128<<<(int)(BN / 16), 256, 0, stream>>>(ctx, k_W, k_b, Km);
  k_dyn<<<B_ * 16, 256, 0, stream>>>(Qm, Km, dynA);
  k_rowgraph<<<(int)BN, 256, 0, stream>>>(dynA, adj, g_alpha, dis);
  k_an<<<(int)BN, 256, 0, stream>>>(dynA, dis, An);

  for (int i = 0; i < NBLK_; ++i) {
    k_temporal<<<(int)BN, 256, 0, stream>>>(h, tW + (size_t)i * S_ * S_, tb + i * S_);
    k_gcnlin<<<(int)BN, 256, 0, stream>>>(h, gW + (size_t)i * H_ * H_, hwT);
    k_spatial<<<B_ * 16, 256, 0, stream>>>(h, hwT, An, gb + i * H_);
    k_ln<<<(int)(BN * S_), 128, 0, stream>>>(h, lng + i * H_, lnb + i * H_);
  }
  k_head<<<(int)BN, 256, 0, stream>>>(h, x_future, fW, fb, hW, hb, out);
}
